// FR_MIL_50697793962566
// MI455X (gfx1250) — matrix operand order repaired and checked by reading, not compiled, not run on hardware
//
#include <hip/hip_runtime.h>
#include <math.h>

typedef __attribute__((ext_vector_type(16))) _Float16 v16h;
typedef __attribute__((ext_vector_type(4)))  _Float16 v4h;
typedef __attribute__((ext_vector_type(8)))  float    v8f;

#define BATCH 4
#define SEQ   8192
#define INDIM 1024
#define HID   512
#define HEADS 8
#define HDIM  64
#define TOPK  16
#define NCLS  2
#define ROWS  (BATCH*SEQ)   /* 32768 */
#define HH    91
#define NPIX  (HH*HH)       /* 8281 */
#define ATT   (NPIX+1)      /* 8282 */

#define KTILES (INDIM/32)   /* 32 */
#define CTILES (HID/16)     /* 32 */

// ---------------------------------------------------------------------------
// K0: pre-pack W_feat (f32, INDIM x HID) into f16 B-fragment order:
//     Wpack[kt][ct][lane][e], e contiguous -> each lane's fragment is one
//     contiguous 32-byte v16h load inside the GEMM.
//     B 32x16 order used here (contiguous k, same as A): N = lane&15, K = e + 16*(lane>=16).
// ---------------------------------------------------------------------------
__global__ __launch_bounds__(256) void packB_kernel(
    const float* __restrict__ Wf, _Float16* __restrict__ Wp)
{
  int idx  = blockIdx.x * 256 + threadIdx.x;   // 32768 total
  int lane = idx & 31;
  int ct   = (idx >> 5) & 31;
  int kt   = idx >> 10;
  int n    = ct * 16 + (lane & 15);
  int kk   = kt * 32 + ((lane >> 4) * 16);
  _Float16* dst = Wp + (size_t)idx * 16;
#pragma unroll
  for (int e = 0; e < 16; ++e)
    dst[e] = (_Float16)Wf[(size_t)(kk + e) * HID + n];
}

// ---------------------------------------------------------------------------
// K1: x = inputs @ W_feat + b_feat via WMMA f16 (f32 accumulate).
// Block = 256 threads (8 wave32s). Block tile: 32 rows x 512 cols (full N),
// so the 134MB A matrix is streamed from HBM exactly once.
// Each wave owns 32x64 = 2 M-subtiles x 4 N-subtiles = 8 WMMAs per K-step.
// LDS A tile is double-buffered -> one barrier per K-step, staging for the
// next tile is issued before the fragment loads of the current one.
// ---------------------------------------------------------------------------
__global__ __launch_bounds__(256) void gemm_feat_wmma(
    const float* __restrict__ inp,      // ROWS x INDIM
    const _Float16* __restrict__ Wp,    // packed B fragments
    const float* __restrict__ bf,       // HID
    float* __restrict__ xout)           // ROWS x HID
{
  __shared__ v4h As4[2][32 * 8];        // 2 x (32 rows x 32 halves) = 2x2KB

  const int mBase = blockIdx.x * 32;
  const int wave  = threadIdx.x >> 5;   // 0..7 -> 64-column strip
  const int lane  = threadIdx.x & 31;

  const v16h* Bp = (const v16h*)Wp;     // [kt][ct][lane]

  // staging coords: one float4 per thread per tile (32 rows x 8 float4)
  const int sr  = threadIdx.x >> 3;         // 0..31
  const int sc4 = (threadIdx.x & 7) * 4;    // 0,4,...,28
  const float* srcRow = inp + (size_t)(mBase + sr) * INDIM + sc4;

  v8f acc[2][4] = {};

  // prologue: stage tile kt=0 into buffer 0
  {
    const float4 v = *reinterpret_cast<const float4*>(srcRow);
    v4h h;
    h[0] = (_Float16)v.x; h[1] = (_Float16)v.y;
    h[2] = (_Float16)v.z; h[3] = (_Float16)v.w;
    As4[0][(sr * 32 + sc4) >> 2] = h;
  }
  __syncthreads();

  const int mrow = lane & 15;
  const int koff = (lane >> 4) * 16;

  for (int kt = 0; kt < KTILES; ++kt) {
    const int cur = kt & 1;

    // stage next tile into the other buffer (issued early to hide latency)
    if (kt + 1 < KTILES) {
      const float4 v = *reinterpret_cast<const float4*>(srcRow + (kt + 1) * 32);
      v4h h;
      h[0] = (_Float16)v.x; h[1] = (_Float16)v.y;
      h[2] = (_Float16)v.z; h[3] = (_Float16)v.w;
      As4[cur ^ 1][(sr * 32 + sc4) >> 2] = h;
    }

    // A fragments: one contiguous 16-half run, K = e + 16*(lane>=16) (matches packed B)
    const _Float16* Ab = (const _Float16*)As4[cur];
    v16h a[2];
#pragma unroll
    for (int s = 0; s < 2; ++s) {
      const _Float16* base = Ab + (s * 16 + mrow) * 32;
#pragma unroll
      for (int e = 0; e < 16; ++e) {
        int k = e + koff;
        a[s][e] = base[k];
      }
    }

    // 4 N-subtiles: one contiguous v16h B load each, 2 WMMAs per B fragment
#pragma unroll
    for (int j = 0; j < 4; ++j) {
      v16h bfr = Bp[((size_t)kt * CTILES + (wave * 4 + j)) * 32 + lane];
      acc[0][j] = __builtin_amdgcn_wmma_f32_16x16x32_f16(
          false, a[0], false, bfr, (short)0, acc[0][j], false, false);
      acc[1][j] = __builtin_amdgcn_wmma_f32_16x16x32_f16(
          false, a[1], false, bfr, (short)0, acc[1][j], false, false);
    }

    __syncthreads();   // next buffer staged + current reads done
  }

  // epilogue: D layout VGPR r -> M = r + 8*(lane>=16), N = lane&15
  const int mAdd = (lane >> 4) * 8;
#pragma unroll
  for (int s = 0; s < 2; ++s) {
#pragma unroll
    for (int j = 0; j < 4; ++j) {
      int n = wave * 64 + j * 16 + (lane & 15);
      float bias = bf[n];
#pragma unroll
      for (int r = 0; r < 8; ++r) {
        int m = mBase + s * 16 + r + mAdd;
        xout[(size_t)m * HID + n] = acc[s][j][r] + bias;
      }
    }
  }
}

// ---------------------------------------------------------------------------
// K2: scores = sigmoid(x @ w_enc + b_enc), one wave per row
// ---------------------------------------------------------------------------
__global__ __launch_bounds__(256) void scores_kernel(
    const float* __restrict__ x, const float* __restrict__ w_enc,
    const float* __restrict__ b_enc, float* __restrict__ scores)
{
  int row  = blockIdx.x * 8 + (threadIdx.x >> 5);
  int lane = threadIdx.x & 31;
  if (row >= ROWS) return;
  float s = 0.f;
  const float* xr = x + (size_t)row * HID;
  for (int c = lane; c < HID; c += 32) s += xr[c] * w_enc[c];
#pragma unroll
  for (int off = 16; off; off >>= 1) s += __shfl_xor(s, off, 32);
  if (lane == 0) scores[row] = 1.f / (1.f + expf(-(s + b_enc[0])));
}

// ---------------------------------------------------------------------------
// K3: top-16 per batch (iterative argmax on LDS copy) + Q = mean of top rows
// ---------------------------------------------------------------------------
__global__ __launch_bounds__(256) void topk_q_kernel(
    const float* __restrict__ scores, const float* __restrict__ x,
    float* __restrict__ Qout)
{
  int b = blockIdx.x;
  __shared__ float sc[SEQ];
  __shared__ int   sidx[TOPK];
  __shared__ float redv[256];
  __shared__ int   redi[256];

  for (int i = threadIdx.x; i < SEQ; i += 256) sc[i] = scores[b * SEQ + i];
  __syncthreads();

  for (int k = 0; k < TOPK; ++k) {
    float best = -1e30f; int bi = SEQ;
    for (int i = threadIdx.x; i < SEQ; i += 256) {
      float v = sc[i];
      if (v > best || (v == best && i < bi)) { best = v; bi = i; }
    }
    redv[threadIdx.x] = best; redi[threadIdx.x] = bi;
    __syncthreads();
    for (int s = 128; s; s >>= 1) {
      if (threadIdx.x < s) {
        float v2 = redv[threadIdx.x + s]; int i2 = redi[threadIdx.x + s];
        if (v2 > redv[threadIdx.x] ||
            (v2 == redv[threadIdx.x] && i2 < redi[threadIdx.x])) {
          redv[threadIdx.x] = v2; redi[threadIdx.x] = i2;
        }
      }
      __syncthreads();
    }
    if (threadIdx.x == 0) { sidx[k] = redi[0]; sc[redi[0]] = -1e30f; }
    __syncthreads();
  }

  for (int c = threadIdx.x; c < HID; c += 256) {
    float s = 0.f;
    for (int k = 0; k < TOPK; ++k)
      s += x[((size_t)b * SEQ + sidx[k]) * HID + c];
    Qout[b * HID + c] = s * (1.f / (float)TOPK);
  }
}

// ---------------------------------------------------------------------------
// K4: q = Q @ Wq + bq
// ---------------------------------------------------------------------------
__global__ __launch_bounds__(256) void qproj_kernel(
    const float* __restrict__ Q, const float* __restrict__ Wq,
    const float* __restrict__ bq, float* __restrict__ qv)
{
  int b = blockIdx.x;
  __shared__ float Qs[HID];
  for (int c = threadIdx.x; c < HID; c += 256) Qs[c] = Q[b * HID + c];
  __syncthreads();
  for (int o = threadIdx.x; o < HID; o += 256) {
    float s = bq[o];
    for (int c = 0; c < HID; ++c) s += Qs[c] * Wq[(size_t)c * HID + o];
    qv[b * HID + o] = s;
  }
}

// ---------------------------------------------------------------------------
// K5: fold K-projection into the single query:
//   keff[b,c,h] = sum_d Wk[c, h*64+d] * q[b, h*64+d];  kb[b,h] = bk_h . q_h
// ---------------------------------------------------------------------------
__global__ __launch_bounds__(256) void keff_kernel(
    const float* __restrict__ qv, const float* __restrict__ Wk,
    const float* __restrict__ bk, float* __restrict__ keff,
    float* __restrict__ kb)
{
  int b = blockIdx.x;
  __shared__ float qs[HID];
  for (int c = threadIdx.x; c < HID; c += 256) qs[c] = qv[b * HID + c];
  __syncthreads();
  for (int idx = threadIdx.x; idx < HID * HEADS; idx += 256) {
    int c = idx >> 3, h = idx & 7;
    float s = 0.f;
    for (int d = 0; d < HDIM; ++d)
      s += Wk[(size_t)c * HID + h * HDIM + d] * qs[h * HDIM + d];
    keff[((size_t)b * HID + c) * HEADS + h] = s;
  }
  if (threadIdx.x < HEADS) {
    int h = threadIdx.x;
    float s = 0.f;
    for (int d = 0; d < HDIM; ++d) s += bk[h * HDIM + d] * qs[h * HDIM + d];
    kb[b * HEADS + h] = s;
  }
}

// ---------------------------------------------------------------------------
// K6: z rows = [cls ; depthwise_conv3x3(relu(x-Q)) + bias + relu(x-Q)]
// ---------------------------------------------------------------------------
__global__ __launch_bounds__(256) void convz_kernel(
    const float* __restrict__ x, const float* __restrict__ Q,
    const float* __restrict__ cw, const float* __restrict__ cb,
    const float* __restrict__ cls, float* __restrict__ z)
{
  int b = blockIdx.y;
  int r = blockIdx.x;                       // 0..ATT-1
  float* zr = z + ((size_t)b * ATT + r) * HID;
  if (r == 0) {
    for (int c = threadIdx.x; c < HID; c += 256) zr[c] = cls[c];
    return;
  }
  int n = r - 1;
  int i = n / HH, j = n % HH;
  for (int c = threadIdx.x; c < HID; c += 256) {
    float qvv = Q[b * HID + c];
    float acc = cb[c];
#pragma unroll
    for (int di = 0; di < 3; ++di) {
      int ii = i + di - 1;
      if (ii < 0 || ii >= HH) continue;
#pragma unroll
      for (int dj = 0; dj < 3; ++dj) {
        int jj = j + dj - 1;
        if (jj < 0 || jj >= HH) continue;
        int m   = ii * HH + jj;
        int src = (m < SEQ) ? m : (m - SEQ);
        float v = fmaxf(x[((size_t)b * SEQ + src) * HID + c] - qvv, 0.f);
        acc += cw[c * 9 + di * 3 + dj] * v;
      }
    }
    {   // residual (center)
      int src = (n < SEQ) ? n : (n - SEQ);
      acc += fmaxf(x[((size_t)b * SEQ + src) * HID + c] - qvv, 0.f);
    }
    zr[c] = acc;
  }
}

// ---------------------------------------------------------------------------
// K7: logits[b,h,n] = (z[b,n,:] . keff[b,:,h] + kb[b,h]) / sqrt(HID)
// ---------------------------------------------------------------------------
__global__ __launch_bounds__(256) void logits_kernel(
    const float* __restrict__ z, const float* __restrict__ keff,
    const float* __restrict__ kb, float* __restrict__ logits)
{
  int b    = blockIdx.y;
  int row  = blockIdx.x * 8 + (threadIdx.x >> 5);
  int lane = threadIdx.x & 31;
  if (row >= ATT) return;

  float acc[HEADS];
#pragma unroll
  for (int h = 0; h < HEADS; ++h) acc[h] = 0.f;

  const float* zr = z + ((size_t)b * ATT + row) * HID;
  for (int c = lane; c < HID; c += 32) {
    float zv = zr[c];
    const float* we = keff + ((size_t)b * HID + c) * HEADS;
#pragma unroll
    for (int h = 0; h < HEADS; ++h) acc[h] += zv * we[h];
  }
#pragma unroll
  for (int off = 16; off; off >>= 1)
#pragma unroll
    for (int h = 0; h < HEADS; ++h) acc[h] += __shfl_xor(acc[h], off, 32);

  if (lane == 0) {
    const float scale = 0.04419417382415922f;   // 1/sqrt(512)
#pragma unroll
    for (int h = 0; h < HEADS; ++h)
      logits[((size_t)b * HEADS + h) * ATT + row] =
          (acc[h] + kb[b * HEADS + h]) * scale;
  }
}

// ---------------------------------------------------------------------------
// K8: in-place softmax over the 8282 keys per (b,h)
// ---------------------------------------------------------------------------
__global__ __launch_bounds__(256) void softmax_kernel(float* __restrict__ logits)
{
  int bh = blockIdx.x;              // 0..31
  float* l = logits + (size_t)bh * ATT;
  __shared__ float red[256];

  float mx = -1e30f;
  for (int n = threadIdx.x; n < ATT; n += 256) mx = fmaxf(mx, l[n]);
  red[threadIdx.x] = mx; __syncthreads();
  for (int s = 128; s; s >>= 1) {
    if (threadIdx.x < s) red[threadIdx.x] = fmaxf(red[threadIdx.x], red[threadIdx.x + s]);
    __syncthreads();
  }
  mx = red[0]; __syncthreads();

  float sum = 0.f;
  for (int n = threadIdx.x; n < ATT; n += 256) {
    float e = expf(l[n] - mx); l[n] = e; sum += e;
  }
  red[threadIdx.x] = sum; __syncthreads();
  for (int s = 128; s; s >>= 1) {
    if (threadIdx.x < s) red[threadIdx.x] += red[threadIdx.x + s];
    __syncthreads();
  }
  float inv = 1.f / red[0];
  __syncthreads();
  for (int n = threadIdx.x; n < ATT; n += 256) l[n] *= inv;
}

// ---------------------------------------------------------------------------
// K9: zbar[b,h,:] = sum_n A[b,h,n] * z[b,n,:]   (deterministic, no atomics)
// ---------------------------------------------------------------------------
__global__ __launch_bounds__(256) void zbar_kernel(
    const float* __restrict__ z, const float* __restrict__ A,
    float* __restrict__ zbar)
{
  int bh = blockIdx.x;              // 0..31
  int b  = bh >> 3;
  const float* Ar = A + (size_t)bh * ATT;
  const float* zb = z + (size_t)b * ATT * HID;
  int c0 = threadIdx.x, c1 = threadIdx.x + 256;
  float a0 = 0.f, a1 = 0.f;
  for (int n = 0; n < ATT; ++n) {
    float w = Ar[n];
    const float* zr = zb + (size_t)n * HID;
    a0 += w * zr[c0];
    a1 += w * zr[c1];
  }
  zbar[(size_t)bh * HID + c0] = a0;
  zbar[(size_t)bh * HID + c1] = a1;
}

// ---------------------------------------------------------------------------
// K10: O = q + zbar@Wv + bv;  O2 = O + relu(O@Wo + bo);  out = O2@Wc + bc
// ---------------------------------------------------------------------------
__global__ __launch_bounds__(256) void final_kernel(
    const float* __restrict__ zbar, const float* __restrict__ qv,
    const float* __restrict__ Wv, const float* __restrict__ bv,
    const float* __restrict__ Wo, const float* __restrict__ bo,
    const float* __restrict__ Wc, const float* __restrict__ bc,
    float* __restrict__ out)
{
  int b = blockIdx.x;
  __shared__ float O[HID];
  __shared__ float O2[HID];
  __shared__ float red[256];

  for (int o = threadIdx.x; o < HID; o += 256) {
    int h = o >> 6;
    const float* zb = zbar + ((size_t)b * HEADS + h) * HID;
    float s = 0.f;
    for (int c = 0; c < HID; ++c) s += zb[c] * Wv[(size_t)c * HID + o];
    O[o] = qv[b * HID + o] + s + bv[o];
  }
  __syncthreads();

  for (int o = threadIdx.x; o < HID; o += 256) {
    float s = bo[o];
    for (int c = 0; c < HID; ++c) s += O[c] * Wo[(size_t)c * HID + o];
    O2[o] = O[o] + fmaxf(s, 0.f);
  }
  __syncthreads();

  for (int cl = 0; cl < NCLS; ++cl) {
    float s = 0.f;
    for (int c = threadIdx.x; c < HID; c += 256) s += O2[c] * Wc[c * NCLS + cl];
    red[threadIdx.x] = s; __syncthreads();
    for (int st = 128; st; st >>= 1) {
      if (threadIdx.x < st) red[threadIdx.x] += red[threadIdx.x + st];
      __syncthreads();
    }
    if (threadIdx.x == 0) out[b * NCLS + cl] = red[0] + bc[cl];
    __syncthreads();
  }
}

// ---------------------------------------------------------------------------
extern "C" void kernel_launch(void* const* d_in, const int* in_sizes, int n_in,
                              void* d_out, int out_size, void* d_ws, size_t ws_size,
                              hipStream_t stream)
{
  (void)in_sizes; (void)n_in; (void)out_size; (void)ws_size;

  const float* inputs  = (const float*)d_in[0];
  const float* W_feat  = (const float*)d_in[1];
  const float* b_feat  = (const float*)d_in[2];
  const float* w_enc   = (const float*)d_in[3];
  const float* b_enc   = (const float*)d_in[4];
  const float* cls_tok = (const float*)d_in[5];
  const float* conv_w  = (const float*)d_in[6];
  const float* conv_b  = (const float*)d_in[7];
  const float* Wq      = (const float*)d_in[8];
  const float* bq      = (const float*)d_in[9];
  const float* Wk      = (const float*)d_in[10];
  const float* bk      = (const float*)d_in[11];
  const float* Wv      = (const float*)d_in[12];
  const float* bv      = (const float*)d_in[13];
  const float* Wo      = (const float*)d_in[14];
  const float* bo      = (const float*)d_in[15];
  const float* Wc      = (const float*)d_in[16];
  const float* bc      = (const float*)d_in[17];
  float* out = (float*)d_out;

  float* ws     = (float*)d_ws;
  float* x      = ws;                                   // ROWS*HID
  float* scores = x      + (size_t)ROWS * HID;          // ROWS
  float* Q      = scores + ROWS;                        // BATCH*HID
  float* qv     = Q      + BATCH * HID;                 // BATCH*HID
  float* keff   = qv     + BATCH * HID;                 // BATCH*HID*HEADS
  float* kb     = keff   + (size_t)BATCH * HID * HEADS; // BATCH*HEADS
  float* z      = kb     + BATCH * HEADS;               // BATCH*ATT*HID
  float* logits = z      + (size_t)BATCH * ATT * HID;   // BATCH*HEADS*ATT
  float* zbar   = logits + (size_t)BATCH * HEADS * ATT; // BATCH*HEADS*HID
  _Float16* wpack = (_Float16*)(zbar + BATCH * HEADS * HID); // KT*CT*32*16 halves

  packB_kernel<<<KTILES * CTILES * 32 / 256, 256, 0, stream>>>(W_feat, wpack);
  gemm_feat_wmma<<<ROWS / 32, 256, 0, stream>>>(inputs, wpack, b_feat, x);
  scores_kernel<<<ROWS / 8, 256, 0, stream>>>(x, w_enc, b_enc, scores);
  topk_q_kernel<<<BATCH, 256, 0, stream>>>(scores, x, Q);
  qproj_kernel<<<BATCH, 256, 0, stream>>>(Q, Wq, bq, qv);
  keff_kernel<<<BATCH, 256, 0, stream>>>(qv, Wk, bk, keff, kb);
  convz_kernel<<<dim3(ATT, BATCH), 256, 0, stream>>>(
      x, Q, conv_w, conv_b, cls_tok, z);
  logits_kernel<<<dim3((ATT + 7) / 8, BATCH), 256, 0, stream>>>(
      z, keff, kb, logits);
  softmax_kernel<<<BATCH * HEADS, 256, 0, stream>>>(logits);
  zbar_kernel<<<BATCH * HEADS, 256, 0, stream>>>(z, logits, zbar);
  final_kernel<<<BATCH, 256, 0, stream>>>(
      zbar, qv, Wv, bv, Wo, bo, Wc, bc, out);
}
